// LearnedAttention_39350490366450
// MI455X (gfx1250) — compile-verified, matrix operand order checked by reading, not run on hardware
//
#include <hip/hip_runtime.h>

// ---------------------------------------------------------------------------
// Bahdanau additive attention, fused for MI455X (gfx1250, wave32, WMMA).
// B=32, M=2048, D=H=1024.
//   scores = W_energy . tanh(q@Wq^T + keys@Wk^T)   (fused, pk never stored)
//   attn   = masked softmax(scores)
//   out    = (attn @ values) @ Wk^T                (projection pulled out)
// Hot kernel: 64-row keys tile in LDS (128 KB), 4 WMMAs per B-fragment load.
// ---------------------------------------------------------------------------

#define BATCH 32
#define MMEM  2048
#define DDIM  1024
#define MT    64                                    // m-rows per block

typedef __attribute__((ext_vector_type(16))) __bf16 v16bf;
typedef __attribute__((ext_vector_type(8)))  float  v8f;

union BF16x16 { uint4 q[2]; v16bf v; unsigned short s[16]; };

__device__ __forceinline__ unsigned short f2bf(float f) {
    unsigned int x = __float_as_uint(f);
    unsigned int r = x + 0x7FFFu + ((x >> 16) & 1u);   // round-to-nearest-even
    return (unsigned short)(r >> 16);
}

__device__ __forceinline__ v8f wmma_bf16(v16bf a, v16bf b, v8f c) {
    return __builtin_amdgcn_wmma_f32_16x16x32_bf16(
        /*neg_a=*/false, a, /*neg_b=*/false, b,
        /*c_mod=*/(short)0, c, /*reuse_a=*/false, /*reuse_b=*/false);
}

// One lane's 32 bytes of a 16x32-bf16 fragment.
// lane: m/n = lane&15, khalf = lane>>4 ; chunks at k0+khalf*8 and k0+16+khalf*8.
__device__ __forceinline__ v16bf load_frag(const unsigned short* row, int k0, int khalf) {
    BF16x16 u;
    u.q[0] = *(const uint4*)(row + k0 + khalf * 8);
    u.q[1] = *(const uint4*)(row + k0 + 16 + khalf * 8);
    return u.v;
}

// Build a fragment from fp32 (converted on the fly).
__device__ __forceinline__ v16bf cvt_frag(const float* row, int k0, int khalf) {
    BF16x16 u;
    const float* p0 = row + k0 + khalf * 8;
    const float* p1 = row + k0 + 16 + khalf * 8;
    float4 a = *(const float4*)(p0), b = *(const float4*)(p0 + 4);
    float4 c = *(const float4*)(p1), d = *(const float4*)(p1 + 4);
    u.s[0]=f2bf(a.x); u.s[1]=f2bf(a.y); u.s[2]=f2bf(a.z); u.s[3]=f2bf(a.w);
    u.s[4]=f2bf(b.x); u.s[5]=f2bf(b.y); u.s[6]=f2bf(b.z); u.s[7]=f2bf(b.w);
    u.s[8]=f2bf(c.x); u.s[9]=f2bf(c.y); u.s[10]=f2bf(c.z); u.s[11]=f2bf(c.w);
    u.s[12]=f2bf(d.x); u.s[13]=f2bf(d.y); u.s[14]=f2bf(d.z); u.s[15]=f2bf(d.w);
    return u.v;
}

// --------------------------- kernel 0: prep -------------------------------
__global__ void prep_kernel(const float* __restrict__ Wk, const float* __restrict__ Wq,
                            unsigned short* __restrict__ wkb, unsigned short* __restrict__ wqb,
                            float* __restrict__ wv) {
    int i = blockIdx.x * 256 + threadIdx.x;          // 0 .. 1048575
    wkb[i] = f2bf(Wk[i]);
    wqb[i] = f2bf(Wq[i]);
    if (i < BATCH * DDIM) wv[i] = 0.0f;
}

// ----------------- 32x1024 @ (1024x1024)^T WMMA GEMM ----------------------
// grid (64 h-tiles, 2 m-tiles), 32 threads (one wave).
__global__ void gemm32_kernel(const float* __restrict__ X,           // [32,1024] fp32
                              const unsigned short* __restrict__ Wb, // [1024,1024] bf16
                              float* __restrict__ Y) {               // [32,1024]
    int ht = blockIdx.x, mt = blockIdx.y;
    int lane = threadIdx.x & 31;
    int n = lane & 15, khalf = lane >> 4;
    const float*          xrow = X  + (size_t)(mt * 16 + n) * DDIM;   // A: m = lane&15
    const unsigned short* brow = Wb + (size_t)(ht * 16 + n) * DDIM;   // B: col n = Wk row
    v8f acc = {};
    for (int k0 = 0; k0 < DDIM; k0 += 32) {
        v16bf a = cvt_frag(xrow, k0, khalf);
        v16bf b = load_frag(brow, k0, khalf);
        acc = wmma_bf16(a, b, acc);
    }
    int col = ht * 16 + n;
#pragma unroll
    for (int r = 0; r < 8; ++r) {
        int m = mt * 16 + khalf * 8 + r;
        Y[(size_t)m * DDIM + col] = acc[r];
    }
}

// --------------------- fused scores kernel (hot spot) ---------------------
// grid (M/64, B), 256 threads (8 waves). LDS: 64x1024 bf16 keys tile = 128 KB.
// Each wave owns 8 h-tiles; per k-step: 1 B fragment feeds 4 WMMAs (4 A tiles).
__global__ void __launch_bounds__(256)
scores_kernel(const float* __restrict__ keys,
              const unsigned short* __restrict__ wkb,  // W_key bf16, L2-resident
              const float* __restrict__ pq,            // [B,1024]
              const float* __restrict__ We,            // [1024]
              float* __restrict__ scores) {            // [B,M]
    __shared__ unsigned short sK[MT * DDIM];           // 128 KB
    const int b   = blockIdx.y;
    const int m0  = blockIdx.x * MT;
    const int tid = threadIdx.x;

    // Stage keys tile fp32 -> bf16 into LDS (each thread: 256 elems of one row).
    {
        int row = tid >> 2;                // 64 rows, 4 threads per row
        int c0  = (tid & 3) * 256;
        const float* src = keys + ((size_t)b * MMEM + m0 + row) * DDIM + c0;
        unsigned short* dst = sK + row * DDIM + c0;
#pragma unroll 8
        for (int j = 0; j < 256; j += 4) {
            float4 f = *(const float4*)(src + j);
            unsigned int lo = (unsigned int)f2bf(f.x) | ((unsigned int)f2bf(f.y) << 16);
            unsigned int hi = (unsigned int)f2bf(f.z) | ((unsigned int)f2bf(f.w) << 16);
            *(uint2*)(dst + j) = make_uint2(lo, hi);
        }
    }
    __syncthreads();

    const int w = tid >> 5, lane = tid & 31;
    const int n = lane & 15, khalf = lane >> 4;

    float sacc[4][8];
#pragma unroll
    for (int s = 0; s < 4; ++s)
#pragma unroll
        for (int r = 0; r < 8; ++r) sacc[s][r] = 0.f;

    const unsigned short* arow0 = sK + (n)      * DDIM;   // rows  0..15
    const unsigned short* arow1 = sK + (16 + n) * DDIM;   // rows 16..31
    const unsigned short* arow2 = sK + (32 + n) * DDIM;   // rows 32..47
    const unsigned short* arow3 = sK + (48 + n) * DDIM;   // rows 48..63

    for (int i = 0; i < 8; ++i) {                          // 8 h-tiles per wave
        const int ht = w * 8 + i;
        const int h  = ht * 16 + n;
        const unsigned short* brow = wkb + (size_t)h * DDIM;
        v8f acc0 = {}, acc1 = {}, acc2 = {}, acc3 = {};
#pragma unroll 2
        for (int k0 = 0; k0 < DDIM; k0 += 32) {
            v16bf bf = load_frag(brow, k0, khalf);         // one B frag ...
            if (k0 + 32 < DDIM) __builtin_prefetch(brow + k0 + 32, 0, 1);
            v16bf a0 = load_frag(arow0, k0, khalf);        // ... reused by 4 A tiles
            v16bf a1 = load_frag(arow1, k0, khalf);
            v16bf a2 = load_frag(arow2, k0, khalf);
            v16bf a3 = load_frag(arow3, k0, khalf);
            acc0 = wmma_bf16(a0, bf, acc0);
            acc1 = wmma_bf16(a1, bf, acc1);
            acc2 = wmma_bf16(a2, bf, acc2);
            acc3 = wmma_bf16(a3, bf, acc3);
        }
        const float pqv = pq[b * DDIM + h];                // column-h bias
        const float wev = We[h];
#pragma unroll
        for (int r = 0; r < 8; ++r) {
            sacc[0][r] += wev * tanhf(acc0[r] + pqv);
            sacc[1][r] += wev * tanhf(acc1[r] + pqv);
            sacc[2][r] += wev * tanhf(acc2[r] + pqv);
            sacc[3][r] += wev * tanhf(acc3[r] + pqv);
        }
    }

    // Reduce over the 16 columns (xor<16 stays within each lane half).
#pragma unroll
    for (int off = 1; off < 16; off <<= 1)
#pragma unroll
        for (int s = 0; s < 4; ++s)
#pragma unroll
            for (int r = 0; r < 8; ++r)
                sacc[s][r] += __shfl_xor(sacc[s][r], off, 32);

    __syncthreads();                       // keys tile no longer needed
    float* sS = (float*)sK;                // reuse first 256 B of LDS for 64 scores
    if (tid < MT) sS[tid] = 0.0f;
    __syncthreads();
    if (n == 0) {                          // lanes 0 and 16 of each wave
        const int mb = khalf * 8;
#pragma unroll
        for (int s = 0; s < 4; ++s)
#pragma unroll
            for (int r = 0; r < 8; ++r)
                atomicAdd(&sS[s * 16 + mb + r], sacc[s][r]);
    }
    __syncthreads();
    if (tid < MT) scores[(size_t)b * MMEM + m0 + tid] = sS[tid];
}

// ----------------------- masked softmax per batch -------------------------
__global__ void softmax_kernel(float* __restrict__ attn, const int* __restrict__ lens) {
    __shared__ float red[256];
    const int b = blockIdx.x, tid = threadIdx.x;
    float* sc = attn + (size_t)b * MMEM;
    const int len = lens[b];

    float lm = -INFINITY;
    for (int m = tid; m < MMEM; m += 256) {
        float v = (m < len) ? sc[m] : -INFINITY;
        lm = fmaxf(lm, v);
    }
    red[tid] = lm; __syncthreads();
    for (int s = 128; s > 0; s >>= 1) { if (tid < s) red[tid] = fmaxf(red[tid], red[tid + s]); __syncthreads(); }
    const float mx = red[0]; __syncthreads();

    float ls = 0.f;
    for (int m = tid; m < MMEM; m += 256) {
        float v = (m < len) ? __expf(sc[m] - mx) : 0.f;
        sc[m] = v; ls += v;
    }
    red[tid] = ls; __syncthreads();
    for (int s = 128; s > 0; s >>= 1) { if (tid < s) red[tid] += red[tid + s]; __syncthreads(); }
    const float inv = 1.0f / red[0];
    __syncthreads();
    for (int m = tid; m < MMEM; m += 256) sc[m] *= inv;
}

// -------------------- wv[b,d] = sum_m attn[b,m] * values ------------------
// grid (M/256, D/256, B), 256 threads. Streams all 256 MB of values once.
__global__ void wv_kernel(const float* __restrict__ attn, const float* __restrict__ values,
                          float* __restrict__ wv) {
    __shared__ float sA[256];
    const int b  = blockIdx.z;
    const int m0 = blockIdx.x * 256;
    const int d  = blockIdx.y * 256 + threadIdx.x;
    sA[threadIdx.x] = attn[(size_t)b * MMEM + m0 + threadIdx.x];
    __syncthreads();
    float s = 0.f;
    const float* base = values + ((size_t)b * MMEM + m0) * DDIM + d;
#pragma unroll 4
    for (int j = 0; j < 256; ++j) s += sA[j] * base[(size_t)j * DDIM];
    atomicAdd(&wv[b * DDIM + d], s);
}

// ---------------------------------------------------------------------------
extern "C" void kernel_launch(void* const* d_in, const int* in_sizes, int n_in,
                              void* d_out, int out_size, void* d_ws, size_t ws_size,
                              hipStream_t stream) {
    (void)in_sizes; (void)n_in; (void)out_size; (void)ws_size;
    const float* queries = (const float*)d_in[0];   // [B,1,D]
    const float* keys    = (const float*)d_in[1];   // [B,M,D]
    const float* values  = (const float*)d_in[2];   // [B,M,D]
    const int*   lens    = (const int*)d_in[3];     // [B]
    const float* W_key   = (const float*)d_in[4];   // [D,D]
    const float* W_query = (const float*)d_in[5];   // [D,D]
    const float* W_energy= (const float*)d_in[6];   // [1,D]

    float* out  = (float*)d_out;                    // [B*D] retrieval
    float* attn = out + BATCH * DDIM;               // [B*M] weights (also raw scores)

    char* ws = (char*)d_ws;
    unsigned short* wkb = (unsigned short*)ws;                    // 2 MB  W_key  bf16
    unsigned short* wqb = (unsigned short*)(ws + (2u << 20));     // 2 MB  W_query bf16
    float* pq = (float*)(ws + (4u << 20));                        // 128 KB projected queries
    float* wv = (float*)(ws + (4u << 20) + (128u << 10));         // 128 KB weighted values

    // 1) bf16 weight conversion + zero wv accumulator
    prep_kernel<<<DDIM * DDIM / 256, 256, 0, stream>>>(W_key, W_query, wkb, wqb, wv);
    // 2) pq = queries @ W_query^T   (WMMA)
    gemm32_kernel<<<dim3(DDIM / 16, 2), 32, 0, stream>>>(queries, wqb, pq);
    // 3) fused scores = We . tanh(pq + keys @ W_key^T)   (WMMA hot loop)
    scores_kernel<<<dim3(MMEM / MT, BATCH), 256, 0, stream>>>(keys, wkb, pq, W_energy, attn);
    // 4) masked softmax (in place in d_out weights region)
    softmax_kernel<<<BATCH, 256, 0, stream>>>(attn, lens);
    // 5) wv = attn @ values   (streaming, memory-bound)
    wv_kernel<<<dim3(MMEM / 256, DDIM / 256, BATCH), 256, 0, stream>>>(attn, values, wv);
    // 6) out = wv @ W_key^T   (WMMA)
    gemm32_kernel<<<dim3(DDIM / 16, 2), 32, 0, stream>>>(wv, wkb, out);
}
